// SymmetricTransitionUpBlock_20899310862386
// MI455X (gfx1250) — matrix operand order repaired and checked by reading, not compiled, not run on hardware
//
#include <hip/hip_runtime.h>

// SymmetricTransitionUpBlock for MI455X (gfx1250, wave32, WMMA).
//
// Sizes (fixed by the reference):
#define N1_PTS   262144
#define N2_PTS   65536
#define KNN      16
#define M_PAIRS  (N2_PTS * KNN)   // 1048576
#define C64      64
#define EPS_BN   1e-5f

typedef __attribute__((ext_vector_type(16))) _Float16 v16h;
typedef __attribute__((ext_vector_type(8)))  float    v8f;

__device__ __forceinline__ v8f wmma_f16(v16h a, v16h b, v8f c) {
  // D = A(16x32 f16) x B(32x16 f16) + C(16x16 f32)
  return __builtin_amdgcn_wmma_f32_16x16x32_f16(
      /*neg_a=*/false, a, /*neg_b=*/false, b,
      /*c_mod=*/(short)0, c, /*reuse_a=*/false, /*reuse_b=*/false);
}

// ---------------------------------------------------------------------------
// Kernel 0: zero segment-max (encoded) and segment-sum accumulators.
// ---------------------------------------------------------------------------
__global__ void __launch_bounds__(256)
init_seg_kernel(unsigned* __restrict__ gmax_enc, float* __restrict__ denom) {
  int i = blockIdx.x * 256 + threadIdx.x;
  if (i < N1_PTS) {
    gmax_enc[i] = 0u;   // below any encoded finite float
    denom[i]    = 0.f;
  }
}

// ---------------------------------------------------------------------------
// Kernel 1: dual WMMA GEMM over x2 [N2,64]:
//   up_feat = relu(bn2(x2 @ w2^T + b2))          [N2,64]
//   t2      = x2 @ ws1[:,3:67]^T                 [N2,64]  (raw, constants folded later)
// One wave = 16 output rows, full 64 cols. 8 waves/block -> 128 rows/block.
// ---------------------------------------------------------------------------
__global__ void __launch_bounds__(256)
gemm_x2_dual_kernel(const float* __restrict__ x2,
                    const float* __restrict__ w2,  const float* __restrict__ b2,
                    const float* __restrict__ g2,  const float* __restrict__ be2,
                    const float* __restrict__ m2,  const float* __restrict__ v2,
                    const float* __restrict__ ws1,               // [64,67]
                    float* __restrict__ up_feat, float* __restrict__ t2) {
  const int wave = threadIdx.x >> 5;
  const int lane = threadIdx.x & 31;
  const int hi   = lane >> 4;            // 0: lanes 0-15, 1: lanes 16-31
  const int l15  = lane & 15;
  const int r0   = (blockIdx.x * 8 + wave) * 16;
  const int row  = r0 + l15;             // A-matrix row for this lane

  v8f accU[4], accT[4];
#pragma unroll
  for (int i = 0; i < 4; ++i)
#pragma unroll
    for (int j = 0; j < 8; ++j) { accU[i][j] = 0.f; accT[i][j] = 0.f; }

#pragma unroll
  for (int kc = 0; kc < 64; kc += 32) {
    // A fragment: 16x32 f16: lane<16 holds K kc..kc+15,
    // lane>=16 holds K kc+16..kc+31, row = l15.
    v16h a;
    const float* xr = x2 + (size_t)row * C64 + kc + hi * 16;
#pragma unroll
    for (int j = 0; j < 8; ++j) {
      a[j]     = (_Float16)xr[j];
      a[j + 8] = (_Float16)xr[j + 8];
    }
#pragma unroll
    for (int nt = 0; nt < 4; ++nt) {
      // B fragment: B[k,n] = W[n0+n, kc+k]. Lane<16 holds K kc..kc+15 of col n0+l15,
      // lane>=16 holds K kc+16..kc+31 -> contiguous 16-f32 row segment of W.
      const int col = nt * 16 + l15;
      const int kB  = kc + hi * 16;
      const float* wu = w2  + (size_t)col * C64 + kB;
      const float* wt = ws1 + (size_t)col * 67 + 3 + kB;
      v16h bu, bt;
#pragma unroll
      for (int j = 0; j < 16; ++j) { bu[j] = (_Float16)wu[j]; bt[j] = (_Float16)wt[j]; }
      accU[nt] = wmma_f16(a, bu, accU[nt]);
      accT[nt] = wmma_f16(a, bt, accT[nt]);
    }
  }

  // D layout: VGPR r -> M = r (lanes 0-15) / M = r+8 (lanes 16-31), N = l15.
  const int rbase = r0 + hi * 8;
#pragma unroll
  for (int nt = 0; nt < 4; ++nt) {
    const int cn = nt * 16 + l15;
    const float s = g2[cn] * rsqrtf(v2[cn] + EPS_BN);
    const float t = be2[cn] + (b2[cn] - m2[cn]) * s;
#pragma unroll
    for (int r = 0; r < 8; ++r) {
      up_feat[(size_t)(rbase + r) * C64 + cn] = fmaxf(accU[nt][r] * s + t, 0.f);
      t2[(size_t)(rbase + r) * C64 + cn]      = accT[nt][r];
    }
  }
}

// ---------------------------------------------------------------------------
// Kernel 2: y1 = relu(bn1(x1 @ w1^T + b1)) -> written directly to d_out [N1,64]
// ---------------------------------------------------------------------------
__global__ void __launch_bounds__(256)
gemm_y1_kernel(const float* __restrict__ x1,
               const float* __restrict__ w1, const float* __restrict__ b1,
               const float* __restrict__ g1, const float* __restrict__ be1,
               const float* __restrict__ m1, const float* __restrict__ v1,
               float* __restrict__ out) {
  const int wave = threadIdx.x >> 5;
  const int lane = threadIdx.x & 31;
  const int hi   = lane >> 4;
  const int l15  = lane & 15;
  const int r0   = (blockIdx.x * 8 + wave) * 16;
  const int row  = r0 + l15;

  v8f acc[4];
#pragma unroll
  for (int i = 0; i < 4; ++i)
#pragma unroll
    for (int j = 0; j < 8; ++j) acc[i][j] = 0.f;

#pragma unroll
  for (int kc = 0; kc < 64; kc += 32) {
    v16h a;
    const float* xr = x1 + (size_t)row * C64 + kc + hi * 16;
#pragma unroll
    for (int j = 0; j < 8; ++j) {
      a[j]     = (_Float16)xr[j];
      a[j + 8] = (_Float16)xr[j + 8];
    }
#pragma unroll
    for (int nt = 0; nt < 4; ++nt) {
      const int col = nt * 16 + l15;
      const float* wr = w1 + (size_t)col * C64 + kc + hi * 16;
      v16h b;
#pragma unroll
      for (int j = 0; j < 16; ++j) b[j] = (_Float16)wr[j];
      acc[nt] = wmma_f16(a, b, acc[nt]);
    }
  }

  const int rbase = r0 + hi * 8;
#pragma unroll
  for (int nt = 0; nt < 4; ++nt) {
    const int cn = nt * 16 + l15;
    const float s = g1[cn] * rsqrtf(v1[cn] + EPS_BN);
    const float t = be1[cn] + (b1[cn] - m1[cn]) * s;
#pragma unroll
    for (int r = 0; r < 8; ++r)
      out[(size_t)(rbase + r) * C64 + cn] = fmaxf(acc[nt][r] * s + t, 0.f);
  }
}

// ---------------------------------------------------------------------------
// Kernel 3: per-pair logit + segment-max via flipped-uint atomicMax.
// h_c = relu((t2[n2,c] + p_r . ws1[c,0:3] + bs1[c] - ms[c]) * sc[c] + bes[c])
// logit = sum_c h_c * ws2[c] + bs2
// ---------------------------------------------------------------------------
__global__ void __launch_bounds__(256)
pair_logit_kernel(const float* __restrict__ p1, const float* __restrict__ p2,
                  const int* __restrict__ knn, const float* __restrict__ t2,
                  const float* __restrict__ ws1, const float* __restrict__ bs1,
                  const float* __restrict__ gs,  const float* __restrict__ bes,
                  const float* __restrict__ ms,  const float* __restrict__ vs,
                  const float* __restrict__ ws2, const float* __restrict__ bs2,
                  float* __restrict__ logit, unsigned* __restrict__ gmax_enc) {
  __shared__ float s_sc[C64], s_sh[C64], s_wx[C64], s_wy[C64], s_wz[C64], s_w2[C64];
  const int t = threadIdx.x;
  if (t < C64) {
    const float sc = gs[t] * rsqrtf(vs[t] + EPS_BN);
    s_sc[t] = sc;
    s_sh[t] = bes[t] + (bs1[t] - ms[t]) * sc;
    s_wx[t] = ws1[t * 67 + 0];
    s_wy[t] = ws1[t * 67 + 1];
    s_wz[t] = ws1[t * 67 + 2];
    s_w2[t] = ws2[t];
  }
  __syncthreads();

  const int m   = blockIdx.x * 256 + t;       // pair id
  const int n2  = m >> 4;                     // coarse point
  const int idx = knn[m];                     // dense point
  const float prx = p1[idx * 3 + 0] - p2[n2 * 3 + 0];
  const float pry = p1[idx * 3 + 1] - p2[n2 * 3 + 1];
  const float prz = p1[idx * 3 + 2] - p2[n2 * 3 + 2];
  const float* t2r = t2 + (size_t)n2 * C64;

  float acc = bs2[0];
#pragma unroll 8
  for (int c = 0; c < C64; ++c) {
    float z = t2r[c] + prx * s_wx[c] + pry * s_wy[c] + prz * s_wz[c];
    float h = fmaxf(z * s_sc[c] + s_sh[c], 0.f);
    acc += h * s_w2[c];
  }
  logit[m] = acc;

  // order-preserving float->uint encoding for atomicMax
  unsigned b = __float_as_uint(acc);
  unsigned enc = (b & 0x80000000u) ? ~b : (b | 0x80000000u);
  atomicMax(&gmax_enc[idx], enc);
}

// ---------------------------------------------------------------------------
// Kernel 4: e = exp(logit - gmax[idx]); denom[idx] += e. Overwrites logit with e.
// ---------------------------------------------------------------------------
__global__ void __launch_bounds__(256)
pair_exp_kernel(const int* __restrict__ knn, float* __restrict__ logit,
                const unsigned* __restrict__ gmax_enc, float* __restrict__ denom) {
  const int m   = blockIdx.x * 256 + threadIdx.x;
  const int idx = knn[m];
  const unsigned enc = gmax_enc[idx];
  const unsigned b   = (enc & 0x80000000u) ? (enc & 0x7FFFFFFFu) : ~enc;
  const float gmax = __uint_as_float(b);
  const float e = __expf(logit[m] - gmax);
  logit[m] = e;                       // reuse buffer as e[m]
  atomicAdd(&denom[idx], e);
}

// ---------------------------------------------------------------------------
// Kernel 5: prob = e/denom[idx]; out[idx,:] += up_feat[n2,:] * prob (atomic).
// Lane = channel for coalesced atomics; 4 pairs per 256-thread block.
// ---------------------------------------------------------------------------
__global__ void __launch_bounds__(256)
scatter_up_kernel(const int* __restrict__ knn, const float* __restrict__ e,
                  const float* __restrict__ denom, const float* __restrict__ up_feat,
                  float* __restrict__ out) {
  const int c   = threadIdx.x;                  // 0..63
  const int m   = blockIdx.x * 4 + threadIdx.y; // pair id
  const int n2  = m >> 4;
  const int idx = knn[m];
  const float prob = e[m] / denom[idx];
  atomicAdd(&out[(size_t)idx * C64 + c], up_feat[(size_t)n2 * C64 + c] * prob);
}

// ---------------------------------------------------------------------------
// Host launcher. Input order follows setup_inputs():
// 0:p1 1:p2 2:x1 3:x2 4:knn_idx 5:w1 6:b1 7:g1 8:be1 9:m1 10:v1
// 11:w2 12:b2 13:g2 14:be2 15:m2 16:v2
// 17:ws1 18:bs1 19:gs 20:bes 21:ms 22:vs 23:ws2 24:bs2
// ---------------------------------------------------------------------------
extern "C" void kernel_launch(void* const* d_in, const int* in_sizes, int n_in,
                              void* d_out, int out_size, void* d_ws, size_t ws_size,
                              hipStream_t stream) {
  const float* p1  = (const float*)d_in[0];
  const float* p2  = (const float*)d_in[1];
  const float* x1  = (const float*)d_in[2];
  const float* x2  = (const float*)d_in[3];
  const int*   knn = (const int*)  d_in[4];
  const float* w1  = (const float*)d_in[5];
  const float* b1  = (const float*)d_in[6];
  const float* g1  = (const float*)d_in[7];
  const float* be1 = (const float*)d_in[8];
  const float* m1  = (const float*)d_in[9];
  const float* v1  = (const float*)d_in[10];
  const float* w2  = (const float*)d_in[11];
  const float* b2  = (const float*)d_in[12];
  const float* g2  = (const float*)d_in[13];
  const float* be2 = (const float*)d_in[14];
  const float* m2  = (const float*)d_in[15];
  const float* v2  = (const float*)d_in[16];
  const float* ws1 = (const float*)d_in[17];
  const float* bs1 = (const float*)d_in[18];
  const float* gs  = (const float*)d_in[19];
  const float* bes = (const float*)d_in[20];
  const float* ms  = (const float*)d_in[21];
  const float* vs  = (const float*)d_in[22];
  const float* ws2 = (const float*)d_in[23];
  const float* bs2 = (const float*)d_in[24];
  float* out = (float*)d_out;

  // Workspace layout (~40 MB)
  float*    up_feat  = (float*)d_ws;                        // N2*64
  float*    t2       = up_feat + (size_t)N2_PTS * C64;      // N2*64
  float*    logit_e  = t2 + (size_t)N2_PTS * C64;           // M_PAIRS (logit, then e)
  unsigned* gmax_enc = (unsigned*)(logit_e + M_PAIRS);      // N1
  float*    denom    = (float*)(gmax_enc + N1_PTS);         // N1

  // 0) zero segment accumulators (d_ws is NOT re-poisoned between replays)
  init_seg_kernel<<<dim3(N1_PTS / 256), dim3(256), 0, stream>>>(gmax_enc, denom);

  // 1) up_feat + t2 from x2 (dual WMMA GEMM): 128 rows/block
  gemm_x2_dual_kernel<<<dim3(N2_PTS / 128), dim3(256), 0, stream>>>(
      x2, w2, b2, g2, be2, m2, v2, ws1, up_feat, t2);

  // 2) y1 -> d_out (WMMA GEMM)
  gemm_y1_kernel<<<dim3(N1_PTS / 128), dim3(256), 0, stream>>>(
      x1, w1, b1, g1, be1, m1, v1, out);

  // 3) per-pair logits + segment max
  pair_logit_kernel<<<dim3(M_PAIRS / 256), dim3(256), 0, stream>>>(
      p1, p2, knn, t2, ws1, bs1, gs, bes, ms, vs, ws2, bs2, logit_e, gmax_enc);

  // 4) exp + segment sum
  pair_exp_kernel<<<dim3(M_PAIRS / 256), dim3(256), 0, stream>>>(
      knn, logit_e, gmax_enc, denom);

  // 5) normalized weighted scatter into d_out
  scatter_up_kernel<<<dim3(M_PAIRS / 4), dim3(64, 4), 0, stream>>>(
      knn, logit_e, denom, up_feat, out);
}
